// BaselineModel_6760278524255
// MI455X (gfx1250) — compile-verified, matrix operand order checked by reading, not run on hardware
//
#include <hip/hip_runtime.h>

typedef __attribute__((ext_vector_type(16))) _Float16 v16h;
typedef __attribute__((ext_vector_type(8)))  _Float16 v8h;
typedef __attribute__((ext_vector_type(4)))  _Float16 v4h;
typedef __attribute__((ext_vector_type(8)))  float    v8f;

__device__ __forceinline__ v16h cat8(v8h lo, v8h hi) {
    v16h r;
#pragma unroll
    for (int i = 0; i < 8; ++i) { r[i] = lo[i]; r[8 + i] = hi[i]; }
    return r;
}
__device__ __forceinline__ v8f wmma16(v16h a, v16h b, v8f c) {
    return __builtin_amdgcn_wmma_f32_16x16x32_f16(false, a, false, b, (short)0, c, false, false);
}

// ===========================================================================
// Repack kernels — all division-free
// ===========================================================================
__global__ void cvt_f16(const float* __restrict__ s, _Float16* __restrict__ d, int n)
{
    int i = blockIdx.x * blockDim.x + threadIdx.x;
    if (i < n) d[i] = (_Float16)s[i];
}

// OIHW f32 -> (tap, co, ci) f16 ; grid (Cin/64, Cout, KK), block 64
__global__ void repack_conv_w(const float* __restrict__ s, _Float16* __restrict__ d,
                              int Cin, int Cout, int KK)
{
    const int ci = blockIdx.x * 64 + threadIdx.x;
    const int co = blockIdx.y;
    const int tap = blockIdx.z;
    d[((size_t)tap * Cout + co) * Cin + ci] =
        (_Float16)s[((size_t)co * Cin + ci) * KK + tap];
}

// stem weights (64,3,7,7) -> (tap[56 pad], co, ci[4 pad]) f16, zero fill
__global__ void repack_stem_w(const float* __restrict__ s, _Float16* __restrict__ d)
{
    int t = blockIdx.x * blockDim.x + threadIdx.x;
    if (t >= 56 * 64 * 4) return;
    int ci = t & 3; int q = t >> 2;
    int co = q & 63; int tap = q >> 6;
    d[t] = (tap < 49 && ci < 3) ? (_Float16)s[(co * 3 + ci) * 49 + tap] : (_Float16)0;
}

// img f32 NCHW -> (64,224,224,4) f16 NHWC ; grid (224,64), block 224
__global__ void repack_img(const float* __restrict__ s, _Float16* __restrict__ d)
{
    const int w = threadIdx.x, h = blockIdx.x, b = blockIdx.y;
    const size_t po = ((size_t)(b * 224 + h) * 224 + w) * 4;
    const size_t si = (size_t)b * 3 * 50176 + (size_t)h * 224 + w;
    d[po + 0] = (_Float16)s[si];
    d[po + 1] = (_Float16)s[si + 50176];
    d[po + 2] = (_Float16)s[si + 100352];
    d[po + 3] = (_Float16)0;
}

// ===========================================================================
// Interior conv, f16 NHWC, implicit GEMM; templated on output spatial S.
// Block = 128 threads (4 waves); wave carries TWO pixel-tile accumulators
// sharing one LDS A fragment. Per tap, a wave-uniform interior test picks a
// select-free fast path (pure b128 loads -> wmma) vs the border path.
// ===========================================================================
template <int S>
__global__ __launch_bounds__(128)
void conv_wmma_nhwc(const _Float16* __restrict__ xact, const _Float16* __restrict__ wrep,
                    const float* __restrict__ bng, const float* __restrict__ bnb,
                    const float* __restrict__ bnm, const float* __restrict__ bnv,
                    const _Float16* __restrict__ res, _Float16* __restrict__ out,
                    int Cin, int Hin, int Cout, int KH, int stride, int pad, int relu)
{
    __shared__ __align__(16) _Float16 smem[16 * 512];
    const int tid  = threadIdx.x;
    const int wave = tid >> 5;
    const int lane = tid & 31;
    const int half = lane >> 4, lr = lane & 15, kb = half * 8;
    const int m0 = blockIdx.y * 16;
    const int Ntot = 64 * S * S;

    const int base = (blockIdx.x * 8 + wave * 2) * 16;     // 32 contiguous pixels/wave
    const int nA  = base + lr;
    const int nB  = nA + 16;
    const int nAc = min(nA, Ntot - 1), nBc = min(nB, Ntot - 1);
    const int pbA = nAc / (S * S); const int rA = nAc - pbA * (S * S);
    const int hoA = rA / S,        woA = rA - hoA * S;
    const int pbB = nBc / (S * S); const int rB = nBc - pbB * (S * S);
    const int hoB = rB / S,        woB = rB - hoB * S;
    const int hbA = hoA * stride - pad, wbA = woA * stride - pad;
    const int hbB = hoB * stride - pad, wbB = woB * stride - pad;

    // wave-uniform ho/wo bounds over the 32-pixel range (conservative)
    const int pend = min(base + 31, Ntot - 1);
    int hoMin, hoMax, woMin, woMax;
    {
        const int ba = base / (S * S), bb = pend / (S * S);
        if (ba != bb) { hoMin = 0; hoMax = S - 1; woMin = 0; woMax = S - 1; }
        else {
            const int ra = base - ba * (S * S), rb = pend - bb * (S * S);
            hoMin = ra / S; hoMax = rb / S;
            if (hoMin != hoMax) { woMin = 0; woMax = S - 1; }
            else { woMin = ra - hoMin * S; woMax = rb - hoMax * S; }
        }
    }

    const int CB = Cin >> 5;
    const _Float16* arow = smem + lr * Cin;
    v8f acc0 = {}, acc1 = {};
    int tap = 0;
    for (int dh = 0; dh < KH; ++dh)
    for (int dw = 0; dw < KH; ++dw, ++tap) {
        // ---- stage 16 x Cin weight slice into LDS
        const _Float16* ws = wrep + ((size_t)tap * Cout + m0) * Cin;
        for (int e = tid * 8; e < 16 * Cin; e += 1024)
            *(v8h*)(smem + e) = *(const v8h*)(ws + e);
        __syncthreads();

        const int hiA = hbA + dh, wiA = wbA + dw;
        const int hiB = hbB + dh, wiB = wbB + dw;
        const _Float16* bpA = xact +
            ((size_t)(pbA * Hin + min(max(hiA, 0), Hin - 1)) * Hin + min(max(wiA, 0), Hin - 1)) * Cin;
        const _Float16* bpB = xact +
            ((size_t)(pbB * Hin + min(max(hiB, 0), Hin - 1)) * Hin + min(max(wiB, 0), Hin - 1)) * Cin;

        // wave-uniform interior test (scalar branch; EXEC stays all-ones)
        const bool allv = (hoMin * stride - pad + dh >= 0) && (hoMax * stride - pad + dh < Hin) &&
                          (woMin * stride - pad + dw >= 0) && (woMax * stride - pad + dw < Hin);
        if (allv) {
#pragma unroll 2
            for (int cb = 0; cb < CB; ++cb) {
                const int ko = cb * 32 + kb;
                v16h av = cat8(*(const v8h*)(arow + ko), *(const v8h*)(arow + ko + 16));
                acc0 = wmma16(av, cat8(*(const v8h*)(bpA + ko),
                                       *(const v8h*)(bpA + ko + 16)), acc0);
                acc1 = wmma16(av, cat8(*(const v8h*)(bpB + ko),
                                       *(const v8h*)(bpB + ko + 16)), acc1);
            }
        } else {
            const bool pvA = ((unsigned)hiA < (unsigned)Hin) & ((unsigned)wiA < (unsigned)Hin);
            const bool pvB = ((unsigned)hiB < (unsigned)Hin) & ((unsigned)wiB < (unsigned)Hin);
            const v8h z = {};
#pragma unroll 2
            for (int cb = 0; cb < CB; ++cb) {
                const int ko = cb * 32 + kb;
                v16h av = cat8(*(const v8h*)(arow + ko), *(const v8h*)(arow + ko + 16));
                v8h qA0 = *(const v8h*)(bpA + ko);
                v8h qA1 = *(const v8h*)(bpA + ko + 16);
                v8h qB0 = *(const v8h*)(bpB + ko);
                v8h qB1 = *(const v8h*)(bpB + ko + 16);
                acc0 = wmma16(av, cat8(pvA ? qA0 : z, pvA ? qA1 : z), acc0);
                acc1 = wmma16(av, cat8(pvB ? qB0 : z, pvB ? qB1 : z), acc1);
            }
        }
        __syncthreads();
    }
    // ---- shared BN epilogue for both pixel tiles
    const int cb0 = m0 + half * 8;
    float scl[8], shf[8];
#pragma unroll
    for (int r = 0; r < 8; ++r) {
        const int co = cb0 + r;
        scl[r] = bng[co] * rsqrtf(bnv[co] + 1e-5f);
        shf[r] = bnb[co] - bnm[co] * scl[r];
    }
    auto store = [&](int nn, const v8f& a) {
        if (nn >= Ntot) return;
        const size_t ob = (size_t)nn * Cout + cb0;
        v8h rv = {};
        if (res) rv = *(const v8h*)(res + ob);
        v8h ov;
#pragma unroll
        for (int r = 0; r < 8; ++r) {
            float y = a[r] * scl[r] + shf[r];
            if (res)  y += (float)rv[r];
            if (relu) y = fmaxf(y, 0.f);
            ov[r] = (_Float16)y;
        }
        *(v8h*)(out + ob) = ov;
    };
    store(nA, acc0);
    store(nB, acc1);
}

// ===========================================================================
// Stem conv 7x7 s2 p3, virtual Cin=4, K=224 (56 padded taps). Each fragment
// covers 4 taps -> 4 x b64 loads for A and B (no scalar gathers).
// ===========================================================================
__global__ __launch_bounds__(32)
void stem_wmma(const _Float16* __restrict__ img, const _Float16* __restrict__ wst,
               const float* __restrict__ bng, const float* __restrict__ bnb,
               const float* __restrict__ bnm, const float* __restrict__ bnv,
               _Float16* __restrict__ out)
{
    const int lane = threadIdx.x & 31, half = lane >> 4, lr = lane & 15, kb = half * 8;
    const int m0 = blockIdx.y * 16;
    const int nn = blockIdx.x * 16 + lr;
    const int pb = nn / 12544; const int r2 = nn - pb * 12544;
    const int pho = r2 / 112, pwo = r2 - pho * 112;
    const int hbase = pho * 2 - 3, wbase = pwo * 2 - 3;

    v8f acc = {};
    for (int k0 = 0; k0 < 224; k0 += 32) {
        v16h av, bv;
        const int t0 = (k0 + kb) >> 2;
#pragma unroll
        for (int g = 0; g < 4; ++g) {
            const int tap = t0 + ((g & 1) ? 1 : 0) + ((g & 2) ? 4 : 0);
            const v4h wv = *(const v4h*)(wst + (tap * 64 + m0 + lr) * 4);
            const int dh = tap / 7, dw = tap - dh * 7;
            const int hi = hbase + dh, wi = wbase + dw;
            const bool pv = ((unsigned)hi < 224u) & ((unsigned)wi < 224u) & (tap < 49);
            const v4h iv = *(const v4h*)(img +
                ((pb * 224 + min(max(hi, 0), 223)) * 224 + min(max(wi, 0), 223)) * 4);
            const v4h z4 = {};
            const v4h sv = pv ? iv : z4;
#pragma unroll
            for (int e = 0; e < 4; ++e) { av[g * 4 + e] = wv[e]; bv[g * 4 + e] = sv[e]; }
        }
        acc = wmma16(av, bv, acc);
    }
    const int cb0 = m0 + half * 8;
    v8h ov;
#pragma unroll
    for (int r = 0; r < 8; ++r) {
        const int co = cb0 + r;
        const float scl = bng[co] * rsqrtf(bnv[co] + 1e-5f);
        const float shf = bnb[co] - bnm[co] * scl;
        ov[r] = (_Float16)fmaxf(acc[r] * scl + shf, 0.f);
    }
    *(v8h*)(out + (size_t)nn * 64 + cb0) = ov;
}

// Maxpool 3x3 s2 p1 NHWC f16 ; grid (7,56,64), block 64 (8 wo x 8 ch-groups)
__global__ void maxpool_nhwc(const _Float16* __restrict__ x, _Float16* __restrict__ o)
{
    const int wo = blockIdx.x * 8 + (threadIdx.x >> 3);
    const int cg = threadIdx.x & 7;
    const int ho = blockIdx.y, b = blockIdx.z;
    float mx[8];
#pragma unroll
    for (int e = 0; e < 8; ++e) mx[e] = -3.402823466e38f;
#pragma unroll
    for (int dh = 0; dh < 3; ++dh)
#pragma unroll
        for (int dw = 0; dw < 3; ++dw) {
            const int hi = ho * 2 - 1 + dh, wi = wo * 2 - 1 + dw;
            if ((unsigned)hi < 112u && (unsigned)wi < 112u) {
                v8h v = *(const v8h*)(x + ((size_t)((b * 112 + hi) * 112 + wi)) * 64 + cg * 8);
#pragma unroll
                for (int e = 0; e < 8; ++e) mx[e] = fmaxf(mx[e], (float)v[e]);
            }
        }
    v8h ov;
#pragma unroll
    for (int e = 0; e < 8; ++e) ov[e] = (_Float16)mx[e];
    *(v8h*)(o + ((size_t)((b * 56 + ho) * 56 + wo)) * 64 + cg * 8) = ov;
}

// Adaptive avg pool (64,7,7,512) NHWC -> h0 (64,512)
__global__ void avgpool7(const _Float16* __restrict__ x, _Float16* __restrict__ h)
{
    int t = blockIdx.x * blockDim.x + threadIdx.x;
    if (t >= 64 * 512) return;
    const int b = t >> 9, c = t & 511;
    float s = 0.f;
#pragma unroll
    for (int p = 0; p < 49; ++p) s += (float)x[(size_t)(b * 49 + p) * 512 + c];
    h[t] = (_Float16)(s * (1.f / 49.f));
}

__global__ void gru_init(const float* __restrict__ embed,
                         _Float16* __restrict__ x, _Float16* __restrict__ s0)
{
    int t = blockIdx.x * blockDim.x + threadIdx.x;
    if (t >= 64 * 512) return;
    _Float16 v = (_Float16)embed[512 + (t & 511)];
    x[t] = v; s0[t] = v;
}

// ===========================================================================
// GRU step GEMMs, M=64 N=1536 K=512. One wave = 4 N-tiles sharing the A
// (x-row) fragment; blockIdx.z selects gi vs gh. grid (24,4,2).
// ===========================================================================
__global__ __launch_bounds__(32)
void gru_gemm2(const _Float16* __restrict__ xb, const _Float16* __restrict__ hb,
               const _Float16* __restrict__ Wih, const _Float16* __restrict__ Whh,
               const float* __restrict__ bih, const float* __restrict__ bhh,
               float* __restrict__ gi, float* __restrict__ gh)
{
    const int lane = threadIdx.x & 31, half = lane >> 4, lr = lane & 15, kb = half * 8;
    const int n0 = blockIdx.x * 64;
    const int m0 = blockIdx.y * 16;
    const bool sel = blockIdx.z != 0;
    const _Float16* X    = sel ? hb  : xb;
    const _Float16* Wt   = sel ? Whh : Wih;
    const float*    bias = sel ? bhh : bih;
    float*          out  = sel ? gh  : gi;

    const _Float16* xr = X  + (m0 + lr) * 512;
    const _Float16* wr = Wt + (n0 + lr) * 512;   // tiles at +0, +8192, +16384, +24576
    v8f c0 = {}, c1 = {}, c2 = {}, c3 = {};
#pragma unroll 2
    for (int k0 = 0; k0 < 512; k0 += 32) {
        __builtin_prefetch(wr + k0 + 128, 0, 1);
        const v16h av = cat8(*(const v8h*)(xr + k0 + kb),
                             *(const v8h*)(xr + k0 + kb + 16));
        c0 = wmma16(av, cat8(*(const v8h*)(wr + k0 + kb),
                             *(const v8h*)(wr + k0 + kb + 16)), c0);
        c1 = wmma16(av, cat8(*(const v8h*)(wr + 8192 + k0 + kb),
                             *(const v8h*)(wr + 8192 + k0 + kb + 16)), c1);
        c2 = wmma16(av, cat8(*(const v8h*)(wr + 16384 + k0 + kb),
                             *(const v8h*)(wr + 16384 + k0 + kb + 16)), c2);
        c3 = wmma16(av, cat8(*(const v8h*)(wr + 24576 + k0 + kb),
                             *(const v8h*)(wr + 24576 + k0 + kb + 16)), c3);
    }
    const v8f* cs[4] = { &c0, &c1, &c2, &c3 };
#pragma unroll
    for (int j = 0; j < 4; ++j) {
        const int n = n0 + 16 * j + lr;
        const float bz = bias[n];
#pragma unroll
        for (int r = 0; r < 8; ++r)
            out[(m0 + r + half * 8) * 1536 + n] = (*cs[j])[r] + bz;
    }
}

// GRU gate math with hardware exp; writes new hidden (f16) to h, x, seq[t]
__global__ void gru_gate(const float* __restrict__ gi, const float* __restrict__ gh,
                         _Float16* __restrict__ h, _Float16* __restrict__ x,
                         _Float16* __restrict__ st)
{
    int t = blockIdx.x * blockDim.x + threadIdx.x;
    if (t >= 64 * 512) return;
    const int b = t >> 9, d = t & 511;
    const float* gib = gi + b * 1536;
    const float* ghb = gh + b * 1536;
    const float r  = 1.f / (1.f + __expf(-(gib[d]       + ghb[d])));
    const float z  = 1.f / (1.f + __expf(-(gib[512 + d] + ghb[512 + d])));
    const float a  = gib[1024 + d] + r * ghb[1024 + d];
    const float nn = 1.f - 2.f / (__expf(2.f * a) + 1.f);   // tanh(a)
    const float h2 = (1.f - z) * nn + z * (float)h[t];
    const _Float16 hv = (_Float16)h2;
    h[t] = hv; x[t] = hv; st[t] = hv;
}

// ===========================================================================
// Projection: seq (12864x512 f16) @ pw^T (128x512 f16) + b -> out[b,v,t] f32.
// One wave = 4 M-tiles sharing the B fragment. grid (8, 201).
// ===========================================================================
__global__ __launch_bounds__(32)
void proj_wmma(const _Float16* __restrict__ seq, const _Float16* __restrict__ pw,
               const float* __restrict__ pbias, float* __restrict__ out)
{
    const int lane = threadIdx.x & 31, half = lane >> 4, lr = lane & 15, kb = half * 8;
    const int n0 = blockIdx.x * 16;
    const int mB = blockIdx.y * 64;
    const _Float16* br = pw  + (size_t)(n0 + lr) * 512;
    const _Float16* ar = seq + (size_t)(mB + lr) * 512;  // tiles at +0,+8192,+16384,+24576
    v8f c0 = {}, c1 = {}, c2 = {}, c3 = {};
#pragma unroll 2
    for (int k0 = 0; k0 < 512; k0 += 32) {
        __builtin_prefetch(ar + k0 + 128, 0, 1);
        const v16h bv = cat8(*(const v8h*)(br + k0 + kb),
                             *(const v8h*)(br + k0 + kb + 16));
        c0 = wmma16(cat8(*(const v8h*)(ar + k0 + kb),
                         *(const v8h*)(ar + k0 + kb + 16)), bv, c0);
        c1 = wmma16(cat8(*(const v8h*)(ar + 8192 + k0 + kb),
                         *(const v8h*)(ar + 8192 + k0 + kb + 16)), bv, c1);
        c2 = wmma16(cat8(*(const v8h*)(ar + 16384 + k0 + kb),
                         *(const v8h*)(ar + 16384 + k0 + kb + 16)), bv, c2);
        c3 = wmma16(cat8(*(const v8h*)(ar + 24576 + k0 + kb),
                         *(const v8h*)(ar + 24576 + k0 + kb + 16)), bv, c3);
    }
    const v8f* cs[4] = { &c0, &c1, &c2, &c3 };
    const int v = n0 + lr;
    const float bz = pbias[v];
#pragma unroll
    for (int j = 0; j < 4; ++j) {
#pragma unroll
        for (int r = 0; r < 8; ++r) {
            const int m = mB + 16 * j + r + half * 8;   // t*64 + b
            const int t = m >> 6, b = m & 63;
            out[(b * 128 + v) * 201 + t] = (*cs[j])[r] + bz;
        }
    }
}

// ===========================================================================
// Host driver
// ===========================================================================
extern "C" void kernel_launch(void* const* d_in, const int* in_sizes, int n_in,
                              void* d_out, int out_size, void* d_ws, size_t ws_size,
                              hipStream_t stream)
{
    (void)in_sizes; (void)n_in; (void)out_size; (void)ws_size;
    auto in = [&](int i) { return (const float*)d_in[i]; };

    char* base = (char*)d_ws;
    size_t off = 0;
    auto falloc = [&](size_t n) {
        void* p = base + off; off += (n * 4 + 255) & ~(size_t)255; return (float*)p;
    };
    auto halloc = [&](size_t n) {
        void* p = base + off; off += (n * 2 + 255) & ~(size_t)255; return (_Float16*)p;
    };

    _Float16* img16  = halloc((size_t)64 * 224 * 224 * 4);
    _Float16* stem16 = halloc((size_t)64 * 112 * 112 * 64);
    const size_t ROTE = (size_t)64 * 56 * 56 * 64;
    _Float16* rot[3] = { stem16, stem16 + ROTE, stem16 + 2 * ROTE };
    _Float16* pool16 = halloc(ROTE);
    float*    gi     = falloc(64 * 1536);
    float*    gh     = falloc(64 * 1536);
    _Float16* hbuf   = halloc(64 * 512);
    _Float16* xbuf   = halloc(64 * 512);
    _Float16* seq    = halloc((size_t)201 * 64 * 512);
    _Float16* wst    = halloc(56 * 64 * 4);
    _Float16* wih16  = halloc(1536 * 512);
    _Float16* whh16  = halloc(1536 * 512);
    _Float16* pw16   = halloc(128 * 512);

    struct Blk { int bn1, bn2, bnd, w1, w2, wd; };
    const Blk blocks[4][2] = {
        { { 8, 12, -1, 16, 17, -1 }, { 18, 22, -1, 26, 27, -1 } },
        { { 28, 32, 36, 40, 41, 42 }, { 43, 47, -1, 51, 52, -1 } },
        { { 53, 57, 61, 65, 66, 67 }, { 68, 72, -1, 76, 77, -1 } },
        { { 78, 82, 86, 90, 91, 92 }, { 93, 97, -1, 101, 102, -1 } },
    };

    auto conv = [&](const _Float16* xin, int wi, int bni, const _Float16* res,
                    _Float16* outp, int Cin, int Hin, int Cout, int Hout,
                    int KHW, int stride, int pad, int relu) {
        const int KK = KHW * KHW;
        _Float16* wr = halloc((size_t)Cout * Cin * KK);
        repack_conv_w<<<dim3(Cin / 64, Cout, KK), 64, 0, stream>>>(in(wi), wr, Cin, Cout, KK);
        const int Ntot = 64 * Hout * Hout;
        dim3 g((Ntot + 127) / 128, Cout / 16);
        switch (Hout) {
        case 56: conv_wmma_nhwc<56><<<g, 128, 0, stream>>>(xin, wr, in(bni), in(bni+1), in(bni+2), in(bni+3), res, outp, Cin, Hin, Cout, KHW, stride, pad, relu); break;
        case 28: conv_wmma_nhwc<28><<<g, 128, 0, stream>>>(xin, wr, in(bni), in(bni+1), in(bni+2), in(bni+3), res, outp, Cin, Hin, Cout, KHW, stride, pad, relu); break;
        case 14: conv_wmma_nhwc<14><<<g, 128, 0, stream>>>(xin, wr, in(bni), in(bni+1), in(bni+2), in(bni+3), res, outp, Cin, Hin, Cout, KHW, stride, pad, relu); break;
        default: conv_wmma_nhwc< 7><<<g, 128, 0, stream>>>(xin, wr, in(bni), in(bni+1), in(bni+2), in(bni+3), res, outp, Cin, Hin, Cout, KHW, stride, pad, relu); break;
        }
    };
    auto pick = [&](const void* a, const void* b) -> _Float16* {
        for (int i = 0; i < 3; ++i)
            if (rot[i] != a && rot[i] != b) return rot[i];
        return rot[0];
    };

    // --- input repack + stem + maxpool ------------------------------------
    repack_img<<<dim3(224, 64), 224, 0, stream>>>(in(0), img16);
    repack_stem_w<<<(56 * 64 * 4 + 255) / 256, 256, 0, stream>>>(in(107), wst);
    stem_wmma<<<dim3(802816 / 16, 4), 32, 0, stream>>>(
        img16, wst, in(103), in(104), in(105), in(106), stem16);
    maxpool_nhwc<<<dim3(7, 56, 64), 64, 0, stream>>>(stem16, pool16);

    // --- residual stages ---------------------------------------------------
    const _Float16* x = pool16;
    int Hin = 56, Cin = 64;
    for (int s = 0; s < 4; ++s) {
        const int Cout = 64 << s;
        const int st0  = (s == 0) ? 1 : 2;
        const int Hout = Hin / st0;
        const Blk& b0 = blocks[s][0];
        _Float16* y1 = pick(x, nullptr);
        conv(x, b0.w1, b0.bn1, nullptr, y1, Cin, Hin, Cout, Hout, 3, st0, 1, 1);
        const _Float16* res = x;
        if (b0.wd >= 0) {
            _Float16* sc = pick(x, y1);
            conv(x, b0.wd, b0.bnd, nullptr, sc, Cin, Hin, Cout, Hout, 1, st0, 0, 0);
            res = sc;
        }
        _Float16* y2 = pick(y1, res);
        conv(y1, b0.w2, b0.bn2, res, y2, Cout, Hout, Cout, Hout, 3, 1, 1, 1);
        x = y2;
        const Blk& b1 = blocks[s][1];
        y1 = pick(x, nullptr);
        conv(x, b1.w1, b1.bn1, nullptr, y1, Cout, Hout, Cout, Hout, 3, 1, 1, 1);
        y2 = pick(y1, x);
        conv(y1, b1.w2, b1.bn2, x, y2, Cout, Hout, Cout, Hout, 3, 1, 1, 1);
        x = y2;
        Hin = Hout; Cin = Cout;
    }

    // --- GRU decoder -------------------------------------------------------
    avgpool7<<<128, 256, 0, stream>>>(x, hbuf);
    gru_init<<<128, 256, 0, stream>>>(in(1), xbuf, seq);
    cvt_f16<<<(786432 + 255) / 256, 256, 0, stream>>>(in(3), wih16, 786432);
    cvt_f16<<<(786432 + 255) / 256, 256, 0, stream>>>(in(2), whh16, 786432);
    cvt_f16<<<(65536 + 255) / 256, 256, 0, stream>>>(in(7), pw16, 65536);
    for (int t = 0; t < 200; ++t) {
        gru_gemm2<<<dim3(24, 4, 2), 32, 0, stream>>>(
            xbuf, hbuf, wih16, whh16, in(5), in(4), gi, gh);
        gru_gate<<<128, 256, 0, stream>>>(
            gi, gh, hbuf, xbuf, seq + (size_t)(t + 1) * 32768);
    }

    // --- projection --------------------------------------------------------
    proj_wmma<<<dim3(8, 201), 32, 0, stream>>>(seq, pw16, in(6), (float*)d_out);
}
